// E1_25048249270379
// MI455X (gfx1250) — matrix operand order repaired and checked by reading, not compiled, not run on hardware
//
#include <hip/hip_runtime.h>
#include <hip/hip_bf16.h>
#include <math.h>

#define DIM    768
#define HEADS  12
#define HD     64
#define FFDIM  2304
#define NLAYER 20
#define NTOK   2048
#define NVOCAB 33
#define EPSV   1e-5f
#define CLIPV  8.0f

typedef __attribute__((ext_vector_type(16))) __bf16 v16bf;
typedef __attribute__((ext_vector_type(8)))  float  v8f;

union Frag { v16bf v; uint4 q[2]; unsigned short s[16]; };

__device__ __forceinline__ unsigned short f2bf(float f) {
  union { float f; unsigned u; } x; x.f = f;
  unsigned r = x.u + 0x7FFFu + ((x.u >> 16) & 1u);
  return (unsigned short)(r >> 16);
}
__device__ __forceinline__ float bf2f(unsigned short s) {
  union { unsigned u; float f; } x; x.u = ((unsigned)s) << 16;
  return x.f;
}

// ---------------------------------------------------------------------------
// Tensor Data Mover: async 2D bf16 tile load Global -> LDS (TENSORcnt).
// D# layout per CDNA5 ISA.
// ---------------------------------------------------------------------------
#if defined(__AMDGCN__) && __has_builtin(__builtin_amdgcn_tensor_load_to_lds)
#define USE_TDM 1
#else
#define USE_TDM 0
#endif

#if USE_TDM
typedef __attribute__((ext_vector_type(4))) unsigned int tdm_v4u;
typedef __attribute__((ext_vector_type(8))) int          tdm_v8i;
typedef __attribute__((ext_vector_type(4))) int          tdm_v4i;

__device__ __forceinline__ void tdm_load_tile_2d(
    unsigned lds_addr, const void* gaddr, int tile_d0 /*elems*/, int tile_d1 /*rows*/,
    int row_stride_elems, int tensor_d0, int tensor_d1) {
  const unsigned long long ga = (unsigned long long)(uintptr_t)gaddr;
  tdm_v4u g0;
  g0.x = 1u;                                               // count=1 (valid user D#)
  g0.y = lds_addr;                                         // LDS byte address
  g0.z = (unsigned)(ga & 0xffffffffu);                     // global_addr[31:0]
  g0.w = (unsigned)((ga >> 32) & 0x01ffffffu) | (2u << 30);// global_addr[56:32], type=2
  tdm_v8i g1;
  // data_size=1 (2B), pad_enable=1, pad_interval=3 (16 DWORDs=64B row),
  // pad_amount=3 (4 DWORDs=16B skew)  -> matches As[128][40] layout.
  g1[0] = (int)((1u << 16) | (1u << 20) | (3u << 22) | (3u << 25));
  g1[1] = (int)(((unsigned)tensor_d0 & 0xffffu) << 16);    // tensor_dim0[15:0]
  g1[2] = (int)((((unsigned)tensor_d0 >> 16) & 0xffffu) |
                (((unsigned)tensor_d1 & 0xffffu) << 16));  // dim0[31:16] | dim1[15:0]
  g1[3] = (int)((((unsigned)tensor_d1 >> 16) & 0xffffu) |
                (((unsigned)tile_d0 & 0xffffu) << 16));    // dim1[31:16] | tile_dim0
  g1[4] = (int)((unsigned)tile_d1 & 0xffffu);              // tile_dim1, tile_dim2=0
  g1[5] = (int)row_stride_elems;                           // tensor_dim0_stride[31:0]
  g1[6] = 0;
  g1[7] = 0;
  const tdm_v4i gz = {0, 0, 0, 0};
#if __clang_major__ >= 23
  const tdm_v8i gz8 = {0, 0, 0, 0, 0, 0, 0, 0};
  __builtin_amdgcn_tensor_load_to_lds(g0, g1, gz, gz, gz8, 0);
#else
  __builtin_amdgcn_tensor_load_to_lds(g0, g1, gz, gz, 0);
#endif
}

__device__ __forceinline__ void tdm_wait() {
#if __has_builtin(__builtin_amdgcn_s_wait_tensorcnt)
  __builtin_amdgcn_s_wait_tensorcnt(0);
#else
  asm volatile("s_wait_tensorcnt 0x0" ::: "memory");
#endif
}
#endif  // USE_TDM

// ---------------------------------------------------------------------------
// Generic bf16 WMMA GEMM:  out[m][o] = epilogue( sum_k A[m][k] * W[o][k] )
// A: bf16 row-major [M][K] (TDM-staged to LDS), W: f32 row-major [O][K]
// (VALU-converted to bf16 while staging). Block 128x128, K-step 32; 8 waves,
// each wave 64x32 (4x2 WMMA accumulators).
// ---------------------------------------------------------------------------
enum { M_CLIP_BF16 = 0, M_CLIP_VT = 1, M_F32 = 2, M_RESID_F32 = 3,
       M_BIAS_GELU = 4, M_BIAS_F32 = 5 };

template <int MODE>
__global__ __launch_bounds__(256) void gemm_bf16_kernel(
    const unsigned short* __restrict__ A, const float* __restrict__ W,
    float* __restrict__ outF, unsigned short* __restrict__ outB,
    const float* __restrict__ resid, const float* __restrict__ bias,
    int M, int K, int O) {
  __shared__ unsigned short As[128][40];
  __shared__ unsigned short Bs[128][40];

  const int tid  = threadIdx.x;
  const int lane = tid & 31;
  const int wv   = tid >> 5;
  const int wm   = wv & 1;    // 0..1 : 64-row group
  const int wo   = wv >> 1;   // 0..3 : 32-col group
  const int m0   = blockIdx.y * 128;
  const int o0   = blockIdx.x * 128;
  const int hi   = lane >> 4;
  const int lo   = lane & 15;
  const int lr   = tid >> 1;          // 0..127 staging row
  const int lc   = (tid & 1) << 4;    // 0 or 16 staging col

  v8f acc[4][2] = {};

#if USE_TDM
  // LDS aperture keeps the byte offset in addr[31:0] (ISA 10.2).
  const unsigned lds_as = (unsigned)(uintptr_t)(void*)&As[0][0];
#endif

  for (int kt = 0; kt < K; kt += 32) {
    // Stage W tile (f32 -> bf16) into registers.
    unsigned short bb[16];
    const int orow = o0 + lr;
    if (orow < O) {
      const float4* wg = (const float4*)(W + (size_t)orow * K + kt + lc);
#pragma unroll
      for (int q4 = 0; q4 < 4; ++q4) {
        float4 f = wg[q4];
        bb[q4 * 4 + 0] = f2bf(f.x); bb[q4 * 4 + 1] = f2bf(f.y);
        bb[q4 * 4 + 2] = f2bf(f.z); bb[q4 * 4 + 3] = f2bf(f.w);
      }
    } else {
#pragma unroll
      for (int q4 = 0; q4 < 16; ++q4) bb[q4] = 0;
    }
    if (kt + 32 < K && orow < O)  // -> global_prefetch_b8
      __builtin_prefetch(W + (size_t)orow * K + kt + 32 + lc, 0, 0);
#if !USE_TDM
    const uint4* ag = (const uint4*)(A + (size_t)(m0 + lr) * K + kt + lc);
    uint4 a0 = ag[0], a1 = ag[1];
#endif
    __syncthreads();  // previous-iteration fragment reads complete
#if USE_TDM
    if (wv == 0) {  // wave-uniform: one TDM descriptor for the whole A tile
      tdm_load_tile_2d(lds_as, A + (size_t)m0 * K + kt, 32, 128, K, K, M);
      tdm_wait();
    }
#else
    *(uint4*)&As[lr][lc]     = a0;
    *(uint4*)&As[lr][lc + 8] = a1;
#endif
    *(uint4*)&Bs[lr][lc]     = *(const uint4*)&bb[0];
    *(uint4*)&Bs[lr][lc + 8] = *(const uint4*)&bb[8];
    __syncthreads();

    // ISA A-fragment layout: lane = row (lane&15); chunks K=8*hi.. , 16+8*hi..
    Frag af[4], bf[2];
#pragma unroll
    for (int i = 0; i < 4; ++i) {
      const int row = wm * 64 + i * 16 + lo;
      af[i].q[0] = *(const uint4*)&As[row][8 * hi];
      af[i].q[1] = *(const uint4*)&As[row][16 + 8 * hi];
    }
    // ISA B-fragment layout: lane = col (lane&15); chunks K=8*hi.. , 16+8*hi..
#pragma unroll
    for (int j = 0; j < 2; ++j) {
      const int row = wo * 32 + j * 16 + lo;
      bf[j].q[0] = *(const uint4*)&Bs[row][8 * hi];
      bf[j].q[1] = *(const uint4*)&Bs[row][16 + 8 * hi];
    }
#pragma unroll
    for (int i = 0; i < 4; ++i)
#pragma unroll
      for (int j = 0; j < 2; ++j)
        acc[i][j] = __builtin_amdgcn_wmma_f32_16x16x32_bf16(
            false, af[i].v, false, bf[j].v, (short)0, acc[i][j], false, false);
  }

  // Epilogue. C layout: vgpr r -> row r + 8*hi, col = lane&15.
#pragma unroll
  for (int i = 0; i < 4; ++i) {
    const int mb = m0 + wm * 64 + i * 16 + 8 * hi;
#pragma unroll
    for (int j = 0; j < 2; ++j) {
      const int o = o0 + wo * 32 + j * 16 + lo;
      if (o < O) {
#pragma unroll
        for (int r = 0; r < 8; ++r) {
          const int m = mb + r;
          float v = acc[i][j][r];
          const size_t idx = (size_t)m * O + o;
          if constexpr (MODE == M_CLIP_BF16) {
            v = fminf(fmaxf(v, -CLIPV), CLIPV);
            outB[idx] = f2bf(v);
          } else if constexpr (MODE == M_CLIP_VT) {
            v = fminf(fmaxf(v, -CLIPV), CLIPV);
            outB[(size_t)o * M + m] = f2bf(v);   // transposed V store
          } else if constexpr (MODE == M_F32) {
            outF[idx] = v;
          } else if constexpr (MODE == M_RESID_F32) {
            outF[idx] = resid[idx] + v;
          } else if constexpr (MODE == M_BIAS_GELU) {
            v += bias[o];
            outF[idx] = 0.5f * v * (1.0f + erff(v * 0.70710678118f));
          } else {  // M_BIAS_F32
            outF[idx] = v + bias[o];
          }
        }
      }
    }
  }
}

// ---------------------------------------------------------------------------
// Flash attention: one block per (head, 64-query tile); 4 waves x 16 q-rows.
// Q,K: bf16 [N][DIM] (head-major inner), VT: bf16 [HEADS*HD][N].
// ---------------------------------------------------------------------------
__global__ __launch_bounds__(128) void attn_kernel(
    const unsigned short* __restrict__ Q, const unsigned short* __restrict__ Kb,
    const unsigned short* __restrict__ VT, const int* __restrict__ sids,
    const int* __restrict__ pad, unsigned short* __restrict__ OUT,
    int g, int M) {
  const int h    = blockIdx.x;
  const int q0   = blockIdx.y * 64;
  const int lane = threadIdx.x & 31;
  const int wv   = threadIdx.x >> 5;
  const int hi   = lane >> 4;
  const int lo   = lane & 15;
  __shared__ unsigned short P[4][16][72];

  const int qrow0 = q0 + wv * 16;
  int sid_r[8], pad_r[8];
#pragma unroll
  for (int r = 0; r < 8; ++r) {
    const int m = qrow0 + r + 8 * hi;
    sid_r[r] = sids[m];
    pad_r[r] = pad[m];
  }

  // Q fragments (A operand), K-dim = 64 -> two 32-steps; load once.
  Frag qa[2];
#pragma unroll
  for (int ks = 0; ks < 2; ++ks) {
    const size_t base = (size_t)(qrow0 + lo) * DIM + h * HD + ks * 32 + 8 * hi;
    qa[ks].q[0] = *(const uint4*)(Q + base);
    qa[ks].q[1] = *(const uint4*)(Q + base + 16);
  }

  v8f oacc[4] = {};
  float rmax[8], rsum[8];
#pragma unroll
  for (int r = 0; r < 8; ++r) { rmax[r] = -INFINITY; rsum[r] = 0.f; }

  for (int j0 = 0; j0 < M; j0 += 64) {
    // S = Q @ K^T  (K rows are d-contiguous -> direct B fragments)
    v8f s[4] = {};
#pragma unroll
    for (int nt = 0; nt < 4; ++nt) {
#pragma unroll
      for (int ks = 0; ks < 2; ++ks) {
        Frag kf;
        const size_t base =
            (size_t)(j0 + nt * 16 + lo) * DIM + h * HD + ks * 32 + 8 * hi;
        kf.q[0] = *(const uint4*)(Kb + base);
        kf.q[1] = *(const uint4*)(Kb + base + 16);
        s[nt] = __builtin_amdgcn_wmma_f32_16x16x32_bf16(
            false, qa[ks].v, false, kf.v, (short)0, s[nt], false, false);
      }
    }

    // Mask + scale; collect per-row tile max.
    float pv[4][8], tmax[8];
#pragma unroll
    for (int r = 0; r < 8; ++r) tmax[r] = -INFINITY;
#pragma unroll
    for (int nt = 0; nt < 4; ++nt) {
      const int jc = j0 + nt * 16 + lo;
      const int sj = sids[jc];
      const int pj = pad[jc];
#pragma unroll
      for (int r = 0; r < 8; ++r) {
        const int m = qrow0 + r + 8 * hi;
        const bool intra = (sid_r[r] == sj);
        const bool ok = (g ? (intra || (m >= jc)) : intra) && pad_r[r] && pj;
        const float v = ok ? s[nt][r] * 0.125f : -1e30f;
        pv[nt][r] = v;
        tmax[r] = fmaxf(tmax[r], v);
      }
    }
    // Row reduce across the 16 lanes that share a row (wave32 halves).
#pragma unroll
    for (int r = 0; r < 8; ++r)
#pragma unroll
      for (int off = 1; off < 16; off <<= 1)
        tmax[r] = fmaxf(tmax[r], __shfl_xor(tmax[r], off, 32));

    float fac[8], tsum[8];
#pragma unroll
    for (int r = 0; r < 8; ++r) {
      const float nm = fmaxf(rmax[r], tmax[r]);
      fac[r] = __expf(rmax[r] - nm);
      rmax[r] = nm;
      tsum[r] = 0.f;
    }
#pragma unroll
    for (int nt = 0; nt < 4; ++nt)
#pragma unroll
      for (int r = 0; r < 8; ++r) {
        const float p = __expf(pv[nt][r] - rmax[r]);
        pv[nt][r] = p;
        tsum[r] += p;
      }
#pragma unroll
    for (int r = 0; r < 8; ++r) {
#pragma unroll
      for (int off = 1; off < 16; off <<= 1)
        tsum[r] += __shfl_xor(tsum[r], off, 32);
      rsum[r] = rsum[r] * fac[r] + tsum[r];
    }
#pragma unroll
    for (int dt = 0; dt < 4; ++dt)
#pragma unroll
      for (int r = 0; r < 8; ++r) oacc[dt][r] *= fac[r];

    // P: C-layout -> LDS -> A-layout (per-wave region, DS ops in order).
#pragma unroll
    for (int nt = 0; nt < 4; ++nt)
#pragma unroll
      for (int r = 0; r < 8; ++r)
        P[wv][r + 8 * hi][nt * 16 + lo] = f2bf(pv[nt][r]);

    Frag pa[2];
#pragma unroll
    for (int ks = 0; ks < 2; ++ks) {
      pa[ks].q[0] = *(const uint4*)&P[wv][lo][ks * 32 + 8 * hi];
      pa[ks].q[1] = *(const uint4*)&P[wv][lo][ks * 32 + 16 + 8 * hi];
    }
    // O += P @ V  (VT rows are key-contiguous -> direct B fragments)
#pragma unroll
    for (int dt = 0; dt < 4; ++dt)
#pragma unroll
      for (int ks = 0; ks < 2; ++ks) {
        Frag vf;
        const size_t base =
            (size_t)(h * HD + dt * 16 + lo) * M + j0 + ks * 32 + 8 * hi;
        vf.q[0] = *(const uint4*)(VT + base);
        vf.q[1] = *(const uint4*)(VT + base + 16);
        oacc[dt] = __builtin_amdgcn_wmma_f32_16x16x32_bf16(
            false, pa[ks].v, false, vf.v, (short)0, oacc[dt], false, false);
      }
  }

#pragma unroll
  for (int dt = 0; dt < 4; ++dt)
#pragma unroll
    for (int r = 0; r < 8; ++r) {
      const int m = qrow0 + r + 8 * hi;
      const int d = dt * 16 + lo;
      OUT[(size_t)m * DIM + h * HD + d] = f2bf(oacc[dt][r] / rsum[r]);
    }
}

// ---------------------------------------------------------------------------
// Elementwise / row kernels
// ---------------------------------------------------------------------------
__global__ void embed_kernel(const int* __restrict__ tokens,
                             const int* __restrict__ sids,
                             const float* __restrict__ temb,
                             const float* __restrict__ semb,
                             float* __restrict__ x) {
  const int idx = blockIdx.x * blockDim.x + threadIdx.x;
  const int m = idx / DIM, d = idx % DIM;
  x[idx] = temb[tokens[m] * DIM + d] + semb[sids[m] * DIM + d];
}

__global__ __launch_bounds__(256) void rms_kernel(
    const float* __restrict__ x, const float* __restrict__ w,
    unsigned short* __restrict__ out) {
  const int row = blockIdx.x;
  const float* xr = x + (size_t)row * DIM;
  __shared__ float red[256];
  float s = 0.f;
  for (int i = threadIdx.x; i < DIM; i += 256) { const float v = xr[i]; s += v * v; }
  red[threadIdx.x] = s;
  __syncthreads();
  for (int t = 128; t > 0; t >>= 1) {
    if (threadIdx.x < t) red[threadIdx.x] += red[threadIdx.x + t];
    __syncthreads();
  }
  const float sc = rsqrtf(red[0] / (float)DIM + EPSV);
  unsigned short* orow = out + (size_t)row * DIM;
  for (int i = threadIdx.x; i < DIM; i += 256) orow[i] = f2bf(xr[i] * sc * w[i]);
}

__global__ void rope_kernel(unsigned short* __restrict__ q,
                            unsigned short* __restrict__ k,
                            const int* __restrict__ seqidx,
                            const int* __restrict__ gidx, int g) {
  const int idx = blockIdx.x * blockDim.x + threadIdx.x;  // NTOK*HEADS*32
  const int i  = idx & 31;
  const int hh = (idx >> 5) % HEADS;
  const int m  = idx / (HEADS * 32);
  const int pos = g ? gidx[m] : seqidx[m];
  const float logt = g ? 13.1223633774f : 9.2103403720f;  // ln(500000), ln(10000)
  const float f = (float)pos * __expf(-((float)i / 32.f) * logt);
  float sn, cs;
  __sincosf(f, &sn, &cs);
  const size_t b = (size_t)m * DIM + hh * HD + i;
  const float q1 = bf2f(q[b]), q2 = bf2f(q[b + 32]);
  q[b]      = f2bf(q1 * cs - q2 * sn);
  q[b + 32] = f2bf(q2 * cs + q1 * sn);
  const float k1 = bf2f(k[b]), k2 = bf2f(k[b + 32]);
  k[b]      = f2bf(k1 * cs - k2 * sn);
  k[b + 32] = f2bf(k2 * cs + k1 * sn);
}

__global__ void silugate_kernel(const float* __restrict__ a,
                                const float* __restrict__ b,
                                unsigned short* __restrict__ g, int n) {
  const int i = blockIdx.x * blockDim.x + threadIdx.x;
  if (i < n) {
    const float x = a[i];
    g[i] = f2bf((x / (1.f + __expf(-x))) * b[i]);
  }
}

__global__ __launch_bounds__(256) void lnorm_kernel(
    const float* __restrict__ x, const float* __restrict__ w,
    const float* __restrict__ b, unsigned short* __restrict__ out) {
  const int row = blockIdx.x;
  const float* xr = x + (size_t)row * DIM;
  __shared__ float r1[256], r2[256];
  float s = 0.f, s2 = 0.f;
  for (int i = threadIdx.x; i < DIM; i += 256) { const float v = xr[i]; s += v; s2 += v * v; }
  r1[threadIdx.x] = s; r2[threadIdx.x] = s2;
  __syncthreads();
  for (int t = 128; t > 0; t >>= 1) {
    if (threadIdx.x < t) { r1[threadIdx.x] += r1[threadIdx.x + t]; r2[threadIdx.x] += r2[threadIdx.x + t]; }
    __syncthreads();
  }
  const float mu  = r1[0] / (float)DIM;
  const float var = r2[0] / (float)DIM - mu * mu;
  const float sc  = rsqrtf(var + EPSV);
  unsigned short* orow = out + (size_t)row * DIM;
  for (int i = threadIdx.x; i < DIM; i += 256)
    orow[i] = f2bf((xr[i] - mu) * sc * w[i] + b[i]);
}

// ---------------------------------------------------------------------------
extern "C" void kernel_launch(void* const* d_in, const int* in_sizes, int n_in,
                              void* d_out, int out_size, void* d_ws, size_t ws_size,
                              hipStream_t stream) {
  (void)in_sizes; (void)n_in; (void)out_size; (void)ws_size;
  const int*   tokens = (const int*)d_in[0];
  const int*   seqidx = (const int*)d_in[1];
  const int*   gidx   = (const int*)d_in[2];
  const int*   sids   = (const int*)d_in[3];
  const int*   mpad   = (const int*)d_in[4];
  const float* temb   = (const float*)d_in[5];
  const float* semb   = (const float*)d_in[6];
  const float* anw    = (const float*)d_in[7];
  const float* wq     = (const float*)d_in[8];
  const float* wk     = (const float*)d_in[9];
  const float* wvp    = (const float*)d_in[10];
  const float* wo     = (const float*)d_in[11];
  const float* fnw    = (const float*)d_in[12];
  const float* wa     = (const float*)d_in[13];
  const float* wb     = (const float*)d_in[14];
  const float* wff    = (const float*)d_in[15];
  const float* finw   = (const float*)d_in[16];
  const float* mwin   = (const float*)d_in[17];
  const float* mbin   = (const float*)d_in[18];
  const float* mlnw   = (const float*)d_in[19];
  const float* mlnb   = (const float*)d_in[20];
  const float* mwout  = (const float*)d_in[21];
  const float* mbout  = (const float*)d_in[22];

  char* ws = (char*)d_ws;
  size_t off = 0;
  auto alloc = [&](size_t bytes) -> void* {
    void* p = ws + off;
    off += (bytes + 255) & ~(size_t)255;
    return p;
  };
  float*          X   = (float*)alloc((size_t)NTOK * DIM * 4);
  unsigned short* H   = (unsigned short*)alloc((size_t)NTOK * DIM * 2);
  unsigned short* Qb  = (unsigned short*)alloc((size_t)NTOK * DIM * 2);
  unsigned short* Kbf = (unsigned short*)alloc((size_t)NTOK * DIM * 2);
  unsigned short* VTb = (unsigned short*)alloc((size_t)NTOK * DIM * 2);
  unsigned short* ATT = (unsigned short*)alloc((size_t)NTOK * DIM * 2);
  float*          FA  = (float*)alloc((size_t)NTOK * FFDIM * 4);
  float*          FB  = (float*)alloc((size_t)NTOK * FFDIM * 4);
  unsigned short* Gb  = (unsigned short*)alloc((size_t)NTOK * FFDIM * 2);
  float*          HM  = (float*)alloc((size_t)NTOK * DIM * 4);
  unsigned short* HN  = (unsigned short*)alloc((size_t)NTOK * DIM * 2);

  const dim3 gp768(DIM / 128, NTOK / 128);      // (6,16)
  const dim3 gp2304(FFDIM / 128, NTOK / 128);   // (18,16)
  const dim3 gpOut(1, NTOK / 128);              // (1,16)
  const dim3 gAttn(HEADS, NTOK / 64);           // (12,32)

  embed_kernel<<<(NTOK * DIM) / 256, 256, 0, stream>>>(tokens, sids, temb, semb, X);

  for (int l = 0; l < NLAYER; ++l) {
    const int g = ((l % 3) == 2);
    const size_t wsq = (size_t)l * DIM * DIM;
    const size_t wsf = (size_t)l * FFDIM * DIM;

    rms_kernel<<<NTOK, 256, 0, stream>>>(X, anw + (size_t)l * DIM, H);
    gemm_bf16_kernel<M_CLIP_BF16><<<gp768, 256, 0, stream>>>(
        H, wq + wsq, nullptr, Qb, nullptr, nullptr, NTOK, DIM, DIM);
    gemm_bf16_kernel<M_CLIP_BF16><<<gp768, 256, 0, stream>>>(
        H, wk + wsq, nullptr, Kbf, nullptr, nullptr, NTOK, DIM, DIM);
    gemm_bf16_kernel<M_CLIP_VT><<<gp768, 256, 0, stream>>>(
        H, wvp + wsq, nullptr, VTb, nullptr, nullptr, NTOK, DIM, DIM);
    rope_kernel<<<(NTOK * HEADS * 32) / 256, 256, 0, stream>>>(Qb, Kbf, seqidx, gidx, g);
    attn_kernel<<<gAttn, 128, 0, stream>>>(Qb, Kbf, VTb, sids, mpad, ATT, g, NTOK);
    gemm_bf16_kernel<M_RESID_F32><<<gp768, 256, 0, stream>>>(
        ATT, wo + wsq, X, nullptr, X, nullptr, NTOK, DIM, DIM);

    rms_kernel<<<NTOK, 256, 0, stream>>>(X, fnw + (size_t)l * DIM, H);
    gemm_bf16_kernel<M_F32><<<gp2304, 256, 0, stream>>>(
        H, wa + wsf, FA, nullptr, nullptr, nullptr, NTOK, DIM, FFDIM);
    gemm_bf16_kernel<M_F32><<<gp2304, 256, 0, stream>>>(
        H, wb + wsf, FB, nullptr, nullptr, nullptr, NTOK, DIM, FFDIM);
    silugate_kernel<<<(NTOK * FFDIM) / 256, 256, 0, stream>>>(FA, FB, Gb, NTOK * FFDIM);
    gemm_bf16_kernel<M_RESID_F32><<<gp768, 256, 0, stream>>>(
        Gb, wff + wsf, X, nullptr, X, nullptr, NTOK, FFDIM, DIM);
  }

  rms_kernel<<<NTOK, 256, 0, stream>>>(X, finw, H);
  gemm_bf16_kernel<M_BIAS_GELU><<<gp768, 256, 0, stream>>>(
      H, mwin, HM, nullptr, nullptr, mbin, NTOK, DIM, DIM);
  lnorm_kernel<<<NTOK, 256, 0, stream>>>(HM, mlnw, mlnb, HN);
  gemm_bf16_kernel<M_BIAS_F32><<<gpOut, 256, 0, stream>>>(
      HN, mwout, (float*)d_out, nullptr, nullptr, mbout, NTOK, DIM, NVOCAB);
}
